// HRqVae_18279380812181
// MI455X (gfx1250) — matrix operand order repaired and checked by reading, not compiled, not run on hardware
//
#include <hip/hip_runtime.h>

// ---------------------------------------------------------------------------
// HRQ-VAE forward for MI455X (gfx1250): bf16 WMMA GEMMs + fused RQ argmin.
// Double-buffered LDS staging via GLOBAL_LOAD_ASYNC_TO_LDS_B128: tile k+1 is
// in flight (ASYNCcnt) while tile k feeds the WMMAs; s_wait_asynccnt 4 retires
// exactly the previous tile (async loads complete in order).
// ---------------------------------------------------------------------------

typedef __attribute__((ext_vector_type(16))) __bf16 v16bf;
typedef __attribute__((ext_vector_type(8)))  float  v8f;

union FragU { v16bf v; uint4 q[2]; };

#define B_ROWS 16384
#define IN_DIM 768
#define EMBED  2048
#define KCODES 2048
#define LLEV   3

__device__ __forceinline__ unsigned short f2bf(float f) {
    union { float f; unsigned int u; } c; c.f = f;
    unsigned int u = c.u;
    unsigned int r = u + 0x7FFFu + ((u >> 16) & 1u);   // round-to-nearest-even
    return (unsigned short)(r >> 16);
}
__device__ __forceinline__ float bf2f(unsigned short h) {
    union { unsigned int u; float f; } c; c.u = ((unsigned int)h) << 16;
    return c.f;
}

// ---------------------------------------------------------------------------
// Elementwise f32 -> bf16 (also used for codebooks: already [N=code][K=dim]).
// ---------------------------------------------------------------------------
__global__ void f32_to_bf16_kernel(const float* __restrict__ src,
                                   unsigned short* __restrict__ dst,
                                   long long n) {
    long long i = (long long)blockIdx.x * blockDim.x + threadIdx.x;
    long long stride = (long long)gridDim.x * blockDim.x;
    for (; i < n; i += stride) dst[i] = f2bf(src[i]);
}

// w[Kd][N] f32  ->  wt[N][Kd] bf16   (row per output neuron, contiguous in K)
__global__ __launch_bounds__(256)
void transpose_f32_to_bf16(const float* __restrict__ w,
                           unsigned short* __restrict__ wt, int Kd, int N) {
    __shared__ float tile[32][33];
    const int bx = blockIdx.x * 32;          // N dimension
    const int by = blockIdx.y * 32;          // Kd dimension
    const int tx = threadIdx.x & 31;
    const int ty = threadIdx.x >> 5;         // 0..7
#pragma unroll
    for (int i = 0; i < 4; ++i) {
        int r = by + ty + i * 8;
        tile[ty + i * 8][tx] = w[(size_t)r * N + bx + tx];
    }
    __syncthreads();
#pragma unroll
    for (int i = 0; i < 4; ++i) {
        int r = bx + ty + i * 8;
        wt[(size_t)r * Kd + by + tx] = f2bf(tile[tx][ty + i * 8]);
    }
}

// ||cb[l][k]||^2 for all 3*2048 codes (D = 2048)
__global__ __launch_bounds__(256)
void cbnorm_kernel(const float* __restrict__ cb, float* __restrict__ out) {
    __shared__ float red[256];
    const int blk = blockIdx.x;
    const float* row = cb + (size_t)blk * EMBED;
    float s = 0.f;
    for (int j = threadIdx.x; j < EMBED; j += 256) { float v = row[j]; s += v * v; }
    red[threadIdx.x] = s;
    __syncthreads();
    for (int off = 128; off > 0; off >>= 1) {
        if ((int)threadIdx.x < off) red[threadIdx.x] += red[threadIdx.x + off];
        __syncthreads();
    }
    if (threadIdx.x == 0) out[blk] = red[0];
}

__global__ void init_u64_kernel(unsigned long long* p, int n, unsigned long long v) {
    int i = blockIdx.x * blockDim.x + threadIdx.x;
    if (i < n) p[i] = v;
}

__global__ void fill_bf16_kernel(unsigned short* p, long long n, unsigned short v) {
    long long i = (long long)blockIdx.x * blockDim.x + threadIdx.x;
    long long stride = (long long)gridDim.x * blockDim.x;
    for (; i < n; i += stride) p[i] = v;
}

// quant += cb[id], res -= cb[id]   (one block per row; f32 math, bf16 storage)
__global__ __launch_bounds__(256)
void rq_update_kernel(const float* __restrict__ cb_l,
                      const unsigned long long* __restrict__ rowkey,
                      unsigned short* __restrict__ quant,
                      unsigned short* __restrict__ res) {
    const int b = blockIdx.x;
    const unsigned int k = (unsigned int)(rowkey[b] & 0xFFFFFFFFull);
    const float* e = cb_l + (size_t)k * EMBED;
    const size_t base = (size_t)b * EMBED;
    for (int j = threadIdx.x; j < EMBED; j += 256) {
        float ev = e[j];
        quant[base + j] = f2bf(bf2f(quant[base + j]) + ev);
        res[base + j]   = f2bf(bf2f(res[base + j]) - ev);
    }
}

// ---------------------------------------------------------------------------
// Tiled bf16 WMMA GEMM:  C[M,N] = A[M,Kd] * Wt[N,Kd]^T
//   BM = BN = 128, BK = 32, 256 threads = 8 waves, wave tile = 32x64 (2x4 WMMA)
// Double-buffered async staging: while buf X feeds WMMAs, buf Y is being
// filled by GLOBAL_LOAD_ASYNC_TO_LDS_B128 (4 in flight per wave).
// MODE 0: out bf16 = relu?(C + bias)      MODE 1: out f32 = C + bias
// MODE 2: RQ argmin epilogue: d = cbnorm[col] - 2*C, per-row u64 atomic-min key
// ---------------------------------------------------------------------------
#define LDSW 40   // padded LDS row stride in ushorts (80B: 16B-aligned frags)

template <int MODE, bool RELU>
__global__ __launch_bounds__(256)
void gemm_bf16(const unsigned short* __restrict__ A,   // [M x Kd] bf16 bits
               const unsigned short* __restrict__ Wt,  // [N x Kd] bf16 bits
               const float* __restrict__ bias,         // [N]        (MODE 0/1)
               unsigned short* __restrict__ outb,      //            (MODE 0)
               float* __restrict__ outf,               //            (MODE 1)
               const float* __restrict__ cbnorm,       // [N]        (MODE 2)
               unsigned long long* __restrict__ rowkey,//            (MODE 2)
               int Kd, int N) {
    __shared__ alignas(16) unsigned short sA[2][128 * LDSW];
    __shared__ alignas(16) unsigned short sB[2][128 * LDSW];

    const int t    = threadIdx.x;
    const int lane = t & 31;
    const int w    = t >> 5;        // wave 0..7
    const int wrow = w >> 1;        // 0..3 -> M offset wrow*32
    const int wcol = w & 1;         // 0..1 -> N offset wcol*64
    const int mblk = blockIdx.y * 128;
    const int nblk = blockIdx.x * 128;

    const int ldr = t >> 2;         // staging row 0..63 (x2 passes)
    const int ldc = (t & 3) * 8;    // staging col (ushort units, 16B chunks)

    const unsigned short* Abase = A  + (size_t)mblk * Kd;
    const unsigned short* Bbase = Wt + (size_t)nblk * Kd;

    v8f zero = {};
    v8f acc[2][4];
#pragma unroll
    for (int i = 0; i < 2; ++i)
#pragma unroll
        for (int j = 0; j < 4; ++j) acc[i][j] = zero;

    // issue 4 async global->LDS b128 copies of one (A,B) K-tile into buffer buf
    auto stage = [&](int k0, int buf) {
        const unsigned short* gA0 = &Abase[(size_t)(ldr)      * Kd + k0 + ldc];
        const unsigned short* gA1 = &Abase[(size_t)(ldr + 64) * Kd + k0 + ldc];
        const unsigned short* gB0 = &Bbase[(size_t)(ldr)      * Kd + k0 + ldc];
        const unsigned short* gB1 = &Bbase[(size_t)(ldr + 64) * Kd + k0 + ldc];
        const unsigned lA0 = (unsigned)(uintptr_t)&sA[buf][(ldr)      * LDSW + ldc];
        const unsigned lA1 = (unsigned)(uintptr_t)&sA[buf][(ldr + 64) * LDSW + ldc];
        const unsigned lB0 = (unsigned)(uintptr_t)&sB[buf][(ldr)      * LDSW + ldc];
        const unsigned lB1 = (unsigned)(uintptr_t)&sB[buf][(ldr + 64) * LDSW + ldc];
        asm volatile(
            "global_load_async_to_lds_b128 %0, %4, off\n\t"
            "global_load_async_to_lds_b128 %1, %5, off\n\t"
            "global_load_async_to_lds_b128 %2, %6, off\n\t"
            "global_load_async_to_lds_b128 %3, %7, off"
            :
            : "v"(lA0), "v"(lA1), "v"(lB0), "v"(lB1),
              "v"(gA0), "v"(gA1), "v"(gB0), "v"(gB1)
            : "memory");
    };

    // consume one staged K-tile: 2 A-frags x 4 B-frags -> 8 WMMAs
    auto consume = [&](int buf) {
        const int khalf = lane >> 4;
        const int lrow  = lane & 15;
        FragU a[2], b[4];
#pragma unroll
        for (int i = 0; i < 2; ++i) {
            // A 16x32 bf16: lane half0 -> K {0..7,16..23}, half1 -> {8..15,24..31}
            const unsigned short* p = &sA[buf][(wrow * 32 + i * 16 + lrow) * LDSW];
            a[i].q[0] = *(const uint4*)(p + khalf * 8);
            a[i].q[1] = *(const uint4*)(p + 16 + khalf * 8);
        }
#pragma unroll
        for (int j = 0; j < 4; ++j) {
            // B 32x16 bf16: lane half0 -> K {0..7,16..23}, half1 -> {8..15,24..31}
            const unsigned short* p = &sB[buf][(wcol * 64 + j * 16 + lrow) * LDSW + khalf * 8];
            b[j].q[0] = *(const uint4*)(p);
            b[j].q[1] = *(const uint4*)(p + 16);
        }
#pragma unroll
        for (int i = 0; i < 2; ++i)
#pragma unroll
            for (int j = 0; j < 4; ++j)
                acc[i][j] = __builtin_amdgcn_wmma_f32_16x16x32_bf16(
                    false, a[i].v, false, b[j].v, (short)0, acc[i][j], false, false);
    };

    const int nk = Kd >> 5;          // #K-tiles; even for all shapes here
    stage(0, 0);                     // prologue: tile 0 -> buf 0
    for (int it = 0; it < nk; it += 2) {
        // ---- even step: fetch tile it+1 -> buf1 while consuming buf0 ----
        {
            const int kn = (it + 1 < nk) ? (it + 1) << 5 : it << 5;  // dummy tail
            stage(kn, 1);
            if (it + 2 < nk) {       // L2 prefetch two tiles ahead
                __builtin_prefetch(&Abase[(size_t)ldr * Kd + ((it + 2) << 5) + ldc], 0, 1);
                __builtin_prefetch(&Bbase[(size_t)ldr * Kd + ((it + 2) << 5) + ldc], 0, 1);
            }
            asm volatile("s_wait_asynccnt 0x4" ::: "memory");  // tile `it` landed
            __syncthreads();
            consume(0);
            __syncthreads();
        }
        // ---- odd step: fetch tile it+2 -> buf0 while consuming buf1 ----
        {
            const int kn = (it + 2 < nk) ? (it + 2) << 5 : (it + 1) << 5;
            stage(kn, 0);
            asm volatile("s_wait_asynccnt 0x4" ::: "memory");  // tile `it+1` landed
            __syncthreads();
            consume(1);
            __syncthreads();
        }
    }
    // (up to 4 dummy async loads remain in flight; S_ENDPGM's implicit
    //  wait-idle retires them, and they never alias the epilogue LDS.)

    // ---- epilogue: C layout VGPR e: lanes0-15 -> M=e, lanes16-31 -> M=8+e ----
    const int elrow = lane & 15;
    const int ehalf = lane >> 4;

    if constexpr (MODE == 0 || MODE == 1) {
#pragma unroll
        for (int i = 0; i < 2; ++i) {
#pragma unroll
            for (int j = 0; j < 4; ++j) {
                const int col = nblk + wcol * 64 + j * 16 + elrow;
                const float bv = bias[col];
#pragma unroll
                for (int e = 0; e < 8; ++e) {
                    const int row = mblk + wrow * 32 + i * 16 + ehalf * 8 + e;
                    float v = acc[i][j][e] + bv;
                    if (RELU) v = v > 0.f ? v : 0.f;
                    if constexpr (MODE == 0) outb[(size_t)row * N + col] = f2bf(v);
                    else                     outf[(size_t)row * N + col] = v;
                }
            }
        }
    } else {
        // RQ distance + per-row argmin: d = ||cb||^2 - 2*S (||res||^2 row-const)
        __shared__ float sD[128 * 128];
        __shared__ unsigned long long skey[256];
#pragma unroll
        for (int i = 0; i < 2; ++i)
#pragma unroll
            for (int j = 0; j < 4; ++j) {
                const int cl = wcol * 64 + j * 16 + elrow;
                const float cn = cbnorm[nblk + cl];
#pragma unroll
                for (int e = 0; e < 8; ++e) {
                    const int rl = wrow * 32 + i * 16 + ehalf * 8 + e;
                    sD[rl * 128 + cl] = cn - 2.0f * acc[i][j][e];
                }
            }
        __syncthreads();
        {
            const int r = t >> 1, h = t & 1;
            unsigned long long best = ~0ull;
            for (int c = h * 64; c < h * 64 + 64; ++c) {
                unsigned int u = __float_as_uint(sD[r * 128 + c]);
                u = (u & 0x80000000u) ? ~u : (u | 0x80000000u);  // order-preserving
                unsigned long long key =
                    ((unsigned long long)u << 32) | (unsigned int)(nblk + c);
                best = key < best ? key : best;
            }
            skey[t] = best;
        }
        __syncthreads();
        if (t < 128) {
            unsigned long long a0 = skey[2 * t], a1 = skey[2 * t + 1];
            unsigned long long best = a0 < a1 ? a0 : a1;
            atomicMin(&rowkey[mblk + t], best);     // global_atomic_min_u64
        }
    }
}

// ---------------------------------------------------------------------------
// Host-side orchestration
// ---------------------------------------------------------------------------
extern "C" void kernel_launch(void* const* d_in, const int* in_sizes, int n_in,
                              void* d_out, int out_size, void* d_ws, size_t ws_size,
                              hipStream_t stream) {
    (void)in_sizes; (void)n_in; (void)out_size; (void)ws_size;

    const float* x  = (const float*)d_in[0];
    const float* ew[4] = { (const float*)d_in[1], (const float*)d_in[3],
                           (const float*)d_in[5], (const float*)d_in[7] };
    const float* eb[4] = { (const float*)d_in[2], (const float*)d_in[4],
                           (const float*)d_in[6], (const float*)d_in[8] };
    const float* dw[4] = { (const float*)d_in[9],  (const float*)d_in[11],
                           (const float*)d_in[13], (const float*)d_in[15] };
    const float* db[4] = { (const float*)d_in[10], (const float*)d_in[12],
                           (const float*)d_in[14], (const float*)d_in[16] };
    const float* cb = (const float*)d_in[17];
    float* out = (float*)d_out;

    // layer dims: enc 768->2048->1024->512->2048, dec 2048->512->1024->2048->768
    const int eK[4] = { IN_DIM, 2048, 1024, 512 };
    const int eN[4] = { 2048, 1024, 512, EMBED };
    const int dK[4] = { EMBED, 512, 1024, 2048 };
    const int dN[4] = { 512, 1024, 2048, IN_DIM };

    // ---- workspace carve ----
    size_t off = 0;
    auto take = [&](size_t bytes) -> char* {
        off = (off + 255) & ~(size_t)255;
        char* p = (char*)d_ws + off;
        off += bytes;
        return p;
    };
    unsigned short* xb  = (unsigned short*)take((size_t)B_ROWS * IN_DIM * 2);
    unsigned short* h0  = (unsigned short*)take((size_t)B_ROWS * EMBED * 2);
    unsigned short* h1  = (unsigned short*)take((size_t)B_ROWS * EMBED * 2);
    unsigned short* wtE[4], *wtD[4];
    for (int i = 0; i < 4; ++i) wtE[i] = (unsigned short*)take((size_t)eK[i] * eN[i] * 2);
    for (int i = 0; i < 4; ++i) wtD[i] = (unsigned short*)take((size_t)dK[i] * dN[i] * 2);
    unsigned short* cbb = (unsigned short*)take((size_t)LLEV * KCODES * EMBED * 2);
    float* cbn          = (float*)take((size_t)LLEV * KCODES * 4);
    unsigned long long* rkey = (unsigned long long*)take((size_t)B_ROWS * 8);

    // ---- one-time conversions ----
    f32_to_bf16_kernel<<<2048, 256, 0, stream>>>(x, xb, (long long)B_ROWS * IN_DIM);
    f32_to_bf16_kernel<<<4096, 256, 0, stream>>>(cb, cbb, (long long)LLEV * KCODES * EMBED);
    for (int i = 0; i < 4; ++i)
        transpose_f32_to_bf16<<<dim3(eN[i] / 32, eK[i] / 32), 256, 0, stream>>>(ew[i], wtE[i], eK[i], eN[i]);
    for (int i = 0; i < 4; ++i)
        transpose_f32_to_bf16<<<dim3(dN[i] / 32, dK[i] / 32), 256, 0, stream>>>(dw[i], wtD[i], dK[i], dN[i]);
    cbnorm_kernel<<<LLEV * KCODES, 256, 0, stream>>>(cb, cbn);

    const dim3 blk(256);
    const int MB = B_ROWS / 128;   // 128 row-blocks

    // ---- encoder ----
    gemm_bf16<0, true ><<<dim3(eN[0] / 128, MB), blk, 0, stream>>>(xb, wtE[0], eb[0], h0, nullptr, nullptr, nullptr, eK[0], eN[0]);
    gemm_bf16<0, true ><<<dim3(eN[1] / 128, MB), blk, 0, stream>>>(h0, wtE[1], eb[1], h1, nullptr, nullptr, nullptr, eK[1], eN[1]);
    gemm_bf16<0, true ><<<dim3(eN[2] / 128, MB), blk, 0, stream>>>(h1, wtE[2], eb[2], h0, nullptr, nullptr, nullptr, eK[2], eN[2]);
    gemm_bf16<0, false><<<dim3(eN[3] / 128, MB), blk, 0, stream>>>(h0, wtE[3], eb[3], h1, nullptr, nullptr, nullptr, eK[3], eN[3]); // z -> h1 (res)

    // quant (h0) = 0
    fill_bf16_kernel<<<4096, 256, 0, stream>>>(h0, (long long)B_ROWS * EMBED, (unsigned short)0);

    // ---- residual VQ, 3 levels ----
    for (int l = 0; l < LLEV; ++l) {
        init_u64_kernel<<<(B_ROWS + 255) / 256, 256, 0, stream>>>(rkey, B_ROWS, ~0ull);
        gemm_bf16<2, false><<<dim3(KCODES / 128, MB), blk, 0, stream>>>(
            h1, cbb + (size_t)l * KCODES * EMBED, nullptr, nullptr, nullptr,
            cbn + (size_t)l * KCODES, rkey, EMBED, KCODES);
        rq_update_kernel<<<B_ROWS, 256, 0, stream>>>(
            cb + (size_t)l * KCODES * EMBED, rkey, h0, h1);
    }

    // ---- decoder (h0 = quant) ----
    gemm_bf16<0, true ><<<dim3(dN[0] / 128, MB), blk, 0, stream>>>(h0, wtD[0], db[0], h1, nullptr, nullptr, nullptr, dK[0], dN[0]);
    gemm_bf16<0, true ><<<dim3(dN[1] / 128, MB), blk, 0, stream>>>(h1, wtD[1], db[1], h0, nullptr, nullptr, nullptr, dK[1], dN[1]);
    gemm_bf16<0, true ><<<dim3(dN[2] / 128, MB), blk, 0, stream>>>(h0, wtD[2], db[2], h1, nullptr, nullptr, nullptr, dK[2], dN[2]);
    gemm_bf16<1, false><<<dim3(dN[3] / 128, MB), blk, 0, stream>>>(h1, wtD[3], db[3], nullptr, out, nullptr, nullptr, dK[3], dN[3]);
}
